// StaticLSTM_31473520345496
// MI455X (gfx1250) — matrix operand order repaired and checked by reading, not compiled, not run on hardware
//
#include <hip/hip_runtime.h>
#include <hip/hip_bf16.h>
#include <math.h>

// ---------------------------------------------------------------------------
// StaticLSTM for MI455X (gfx1250):
//   Phase 1: zx = x @ kernel + bias   via v_wmma_f32_16x16x32_bf16
//            (32x64 tile per wave -> 8 WMMA / 12 b128 loads per K-chunk)
//   Phase 2: sequential LSTM scan, persistent kernel, L2-resident R panels
//            (per-block contiguous layout), c-state in registers, grid
//            barrier via global atomics.
// ---------------------------------------------------------------------------

#define T_SEQ 32768
#define D_IN  1024
#define U_HID 1024
#define N4U   4096          // 4*U
#define NBLK_REC 64         // persistent blocks for the recurrence

typedef __attribute__((ext_vector_type(16))) __bf16    v16bf;
typedef __attribute__((ext_vector_type(8)))  float     v8f;
typedef __attribute__((ext_vector_type(4)))  unsigned  u32x4;

union ABfrag { u32x4 q[2]; v16bf v; };   // 32 bytes: one WMMA bf16 A/B fragment

static __device__ __forceinline__ unsigned short f2bf(float f) {
  union { float f; unsigned u; } v; v.f = f;
  unsigned r = v.u + 0x7FFFu + ((v.u >> 16) & 1u);   // round-to-nearest-even
  return (unsigned short)(r >> 16);
}
static __device__ __forceinline__ float bf2f(unsigned short b) {
  union { unsigned u; float f; } v; v.u = ((unsigned)b) << 16;
  return v.f;
}

// ------------------------- conversion kernels ------------------------------

__global__ void conv_x_kernel(const float* __restrict__ x,
                              unsigned short* __restrict__ xb, int n) {
  int i = blockIdx.x * blockDim.x + threadIdx.x;
  if (i < n) xb[i] = f2bf(x[i]);
}

// kernel [K=1024][N=4096] row-major  ->  kT bf16 [N=4096][K=1024]
__global__ void conv_kT_kernel(const float* __restrict__ kmat,
                               unsigned short* __restrict__ kTb) {
  int i = blockIdx.x * blockDim.x + threadIdx.x;   // 0 .. 4096*1024-1
  int nn = i >> 10;            // column of original kernel
  int kk = i & 1023;           // row of original kernel
  kTb[i] = f2bf(kmat[(size_t)kk * N4U + nn]);
}

// recurrent_kernel [K=1024][N=4096] -> per-block panels:
//   rb2[blk][k][c64] , c64 = gate*16 + u_local , blk owns units blk*16..+15.
// Each recurrence block then streams a contiguous 128 KB panel per step and
// every 64-thread group reads one contiguous 128-B line per k.
__global__ void conv_r_kernel(const float* __restrict__ r,
                              unsigned short* __restrict__ rb2) {
  int i = blockIdx.x * blockDim.x + threadIdx.x;   // 0 .. 1024*4096-1 (dst idx)
  int c64 = i & 63;
  int kk  = (i >> 6) & 1023;
  int blk = i >> 16;
  int gate = c64 >> 4, u_local = c64 & 15;
  rb2[i] = f2bf(r[(size_t)kk * N4U + gate * U_HID + blk * 16 + u_local]);
}

__global__ void init_state_kernel(float* __restrict__ h_glob,
                                  unsigned* __restrict__ bar) {
  int i = blockIdx.x * blockDim.x + threadIdx.x;
  if (i < U_HID) h_glob[i] = 0.0f;
  if (i == 0) *bar = 0u;
}

// ------------------------- Phase 1: WMMA GEMM ------------------------------
// Each wave computes a 32(M) x 64(N) output tile; K-loop in chunks of 32.
// Block = 256 threads = 8 waves -> 256 M-rows x 64 N-cols per block.
__global__ __launch_bounds__(256)
void gemm_zx_wmma(const unsigned short* __restrict__ xb,    // [T][D] bf16
                  const unsigned short* __restrict__ kTb,   // [N4U][D] bf16
                  const float* __restrict__ bias,           // [N4U]
                  float* __restrict__ zx) {                 // [T][N4U] f32
  const int lane = threadIdx.x & 31;
  const int wave = threadIdx.x >> 5;
  const int half = lane >> 4;     // 0: lanes 0-15, 1: lanes 16-31
  const int l16  = lane & 15;
  const int m0 = (blockIdx.y * 8 + wave) * 32;
  const int n0 = blockIdx.x * 64;

  v8f acc[2][4];
#pragma unroll
  for (int mt = 0; mt < 2; ++mt)
#pragma unroll
    for (int s = 0; s < 4; ++s) acc[mt][s] = (v8f)(0.0f);

  // A (16x32 bf16): lane half selects K groups [0..7]+[16..23] / [8..15]+[24..31]
  const unsigned short* arow[2];
  arow[0] = xb + (size_t)(m0 + l16) * D_IN + half * 8;
  arow[1] = xb + (size_t)(m0 + 16 + l16) * D_IN + half * 8;
  // B (32x16 bf16): lane l16 = column; half selects K groups [0..7]+[16..23] / [8..15]+[24..31]
  const unsigned short* bp[4];
#pragma unroll
  for (int s = 0; s < 4; ++s)
    bp[s] = kTb + (size_t)(n0 + s * 16 + l16) * D_IN + half * 8;

  for (int kb = 0; kb < D_IN; kb += 32) {
    if (kb + 32 < D_IN) {   // gfx1250 prefetch of next K-chunk (global_prefetch_b8)
      __builtin_prefetch(arow[0] + kb + 32, 0, 1);
      __builtin_prefetch(bp[0] + kb + 32, 0, 1);
    }
    ABfrag A[2];
#pragma unroll
    for (int mt = 0; mt < 2; ++mt) {
      A[mt].q[0] = *(const u32x4*)(arow[mt] + kb);        // K = kb + half*8 + 0..7
      A[mt].q[1] = *(const u32x4*)(arow[mt] + kb + 16);   // K = kb + 16 + half*8 + 0..7
    }
#pragma unroll
    for (int s = 0; s < 4; ++s) {
      ABfrag B;
      B.q[0] = *(const u32x4*)(bp[s] + kb);     // K = kb + half*8 + 0..7
      B.q[1] = *(const u32x4*)(bp[s] + kb + 16); // K = kb + 16 + half*8 + 0..7
#pragma unroll
      for (int mt = 0; mt < 2; ++mt)
        acc[mt][s] = __builtin_amdgcn_wmma_f32_16x16x32_bf16(
            /*neg_a=*/false, A[mt].v, /*neg_b=*/false, B.v,
            /*c_mod=*/(short)0, acc[mt][s], /*reuse_a=*/false, /*reuse_b=*/false);
    }
  }

  // C/D layout: VGPR r -> row M = half*8 + r; col N = l16 (+ subtile offset)
#pragma unroll
  for (int s = 0; s < 4; ++s) {
    const int coln = n0 + s * 16 + l16;
    const float bv = bias[coln];
#pragma unroll
    for (int mt = 0; mt < 2; ++mt)
#pragma unroll
      for (int r = 0; r < 8; ++r)
        zx[(size_t)(m0 + mt * 16 + half * 8 + r) * N4U + coln] = acc[mt][s][r] + bv;
  }
}

// ------------------------- Phase 2: recurrence -----------------------------
// 64 persistent blocks x 256 threads. Block b owns hidden units
// u in [b*16, b*16+16) and all 4 gate columns for them (64 columns).
// part = tid>>6: each 64-thread group handles one 256-long K slice, so a
// group's 64 lanes read one contiguous 128-B line of the R panel per k.
__global__ __launch_bounds__(256)
void lstm_rec(const float* __restrict__ zx,            // [T][N4U] f32
              const unsigned short* __restrict__ rb2,  // [64][1024][64] bf16
              float* __restrict__ h_glob,              // [U] f32 (zero-init)
              float* __restrict__ out,                 // [T][U] f32
              unsigned* __restrict__ bar) {
  __shared__ float h_lds[U_HID];
  __shared__ float partial[4][64];
  __shared__ float zred[64];

  const int tid = threadIdx.x;
  const int blk = blockIdx.x;
  const int col_local = tid & 63;     // 0..63  (contiguous across a warp)
  const int part      = tid >> 6;     // 0..3   (K slice)
  const int gate      = col_local >> 4;   // 0..3  (i, f, g, o)
  const int u_local   = col_local & 15;   // 0..15
  const int col = gate * U_HID + blk * 16 + u_local;  // column of z

  float c = 0.0f;   // meaningful only for tid < 16 (one unit per thread)

  // this thread's streaming pointer into the block's R panel
  const unsigned short* rp =
      rb2 + ((size_t)blk * 1024 + part * 256) * 64 + col_local;

  for (int t = 0; t < T_SEQ; ++t) {
    // broadcast h into LDS (device-scope loads: cross-block coherent)
    for (int i = tid; i < U_HID; i += 256)
      h_lds[i] = __hip_atomic_load(&h_glob[i], __ATOMIC_RELAXED,
                                   __HIP_MEMORY_SCOPE_AGENT);
    __syncthreads();

    // partial dot over a 256-long K slice (R panel hot in L2)
    float acc = 0.0f;
#pragma unroll 8
    for (int k = 0; k < 256; ++k)
      acc = fmaf(h_lds[part * 256 + k], bf2f(rp[(size_t)k * 64]), acc);
    partial[part][col_local] = acc;
    __syncthreads();

    if (part == 0)
      zred[col_local] = zx[(size_t)t * N4U + col] +
                        partial[0][col_local] + partial[1][col_local] +
                        partial[2][col_local] + partial[3][col_local];
    __syncthreads();

    if (tid < 16) {
      const float zi = zred[0 * 16 + tid];
      const float zf = zred[1 * 16 + tid];
      const float zg = zred[2 * 16 + tid];
      const float zo = zred[3 * 16 + tid];
      const float ig = 1.0f / (1.0f + __expf(-zi));
      const float fg = 1.0f / (1.0f + __expf(-zf));
      const float gg = tanhf(zg);
      const float og = 1.0f / (1.0f + __expf(-zo));
      c = fg * c + ig * gg;
      const float h = og * tanhf(c);
      const int uu = blk * 16 + tid;
      out[(size_t)t * U_HID + uu] = h;
      __hip_atomic_store(&h_glob[uu], h, __ATOMIC_RELAXED,
                         __HIP_MEMORY_SCOPE_AGENT);
    }

    // ---- grid barrier (monotonic counter; max 64*32768 < 2^31) ----
    __threadfence();
    __syncthreads();
    if (tid == 0) {
      atomicAdd(bar, 1u);
      const unsigned target = (unsigned)(t + 1) * (unsigned)gridDim.x;
      while (__hip_atomic_load(bar, __ATOMIC_ACQUIRE,
                               __HIP_MEMORY_SCOPE_AGENT) < target) {
        __builtin_amdgcn_s_sleep(1);
      }
    }
    __syncthreads();
  }
}

// ---------------------------------------------------------------------------

extern "C" void kernel_launch(void* const* d_in, const int* in_sizes, int n_in,
                              void* d_out, int out_size, void* d_ws, size_t ws_size,
                              hipStream_t stream) {
  const float* x    = (const float*)d_in[0];   // [1, T, D]
  const float* km   = (const float*)d_in[1];   // [D, 4U]
  const float* rk   = (const float*)d_in[2];   // [U, 4U]
  const float* bias = (const float*)d_in[3];   // [4U]
  float* out = (float*)d_out;                  // [1, T, U]

  char* ws = (char*)d_ws;
  size_t off = 0;
  unsigned short* xb  = (unsigned short*)(ws + off); off += (size_t)T_SEQ * D_IN * 2;  // 64 MB
  unsigned short* kTb = (unsigned short*)(ws + off); off += (size_t)D_IN  * N4U * 2;   //  8 MB
  unsigned short* rb2 = (unsigned short*)(ws + off); off += (size_t)U_HID * N4U * 2;   //  8 MB
  float*    zx     = (float*)(ws + off);             off += (size_t)T_SEQ * N4U * 4;   // 512 MB
  float*    h_glob = (float*)(ws + off);             off += (size_t)U_HID * 4;
  off = (off + 63) & ~(size_t)63;
  unsigned* bar    = (unsigned*)(ws + off);          off += 64;
  (void)ws_size; (void)in_sizes; (void)n_in; (void)out_size;

  conv_x_kernel <<<(T_SEQ * D_IN) / 256, 256, 0, stream>>>(x, xb, T_SEQ * D_IN);
  conv_kT_kernel<<<(D_IN * N4U) / 256, 256, 0, stream>>>(km, kTb);
  conv_r_kernel <<<(U_HID * N4U) / 256, 256, 0, stream>>>(rk, rb2);
  init_state_kernel<<<4, 256, 0, stream>>>(h_glob, bar);

  dim3 grid(N4U / 64, T_SEQ / 256);   // (64, 128) blocks of 8 waves
  gemm_zx_wmma<<<grid, 256, 0, stream>>>(xb, kTb, bias, zx);

  lstm_rec<<<NBLK_REC, 256, 0, stream>>>(zx, rb2, h_glob, out, bar);
}
